// MHA_62569083568181
// MI455X (gfx1250) — matrix operand order repaired and checked by reading, not compiled, not run on hardware
//
#include <hip/hip_runtime.h>

// ---------------------------------------------------------------------------
// MHA forward for gfx1250 (MI455X): bf16 WMMA + TDM (tensor_load_to_lds).
// B=2, S=2048, D=1024, H=16, dh=64.
//
// Pipeline:
//   0a. W[k][n] fp32 -> WT[n][k] bf16 (LDS-tiled transpose)   x4
//   0b. X fp32 -> X bf16 (streaming, v_cvt_pk_bf16_f32)       x3
//   1.  Q/K projection  (transposed orientation, b128 epilogue, Q pre-scaled)
//   2.  V projection    (natural orientation -> V^T layout [b][h][dh][s])
//   3.  flash attention (transposed scores; TDM double-buffers K/V tiles into
//       LDS tracked by TENSORcnt; per-lane softmax stats + 1 shfl; P via LDS)
//   4.  output projection (bf16 ctx @ Wo^T -> fp32 + bias)
// ---------------------------------------------------------------------------

#define B_  2
#define S_  2048
#define D_  1024
#define H_  16
#define DH_ 64
#define T_  (B_*S_)   // 4096 tokens

typedef __attribute__((ext_vector_type(16))) __bf16 v16bf;
typedef __attribute__((ext_vector_type(8)))  float  v8f;
typedef __attribute__((ext_vector_type(4)))  unsigned int v4u;
typedef __attribute__((ext_vector_type(8)))  int    v8i;
typedef __attribute__((ext_vector_type(4)))  int    v4i;

union FragU { uint4 u[2]; v16bf v; };
union Pack8 { __bf16 h[8]; uint4 u; };

__device__ __forceinline__ v8f wmma_bf16(v16bf a, v16bf b, v8f c) {
  // D = A(16x32 bf16) * B(32x16 bf16) + C(16x16 f32)
  return __builtin_amdgcn_wmma_f32_16x16x32_bf16(false, a, false, b,
                                                 (short)0, c, false, false);
}

// A operand (16x32 bf16, M x K) from row-major bf16 [rows][ld].
__device__ __forceinline__ v16bf load_a_bf16(const __bf16* base, int row0,
                                             int ld, int k0, int lane) {
  const int sel = (lane >> 4) & 1;
  const __bf16* p = base + (size_t)(row0 + (lane & 15)) * ld + k0 + 16 * sel;
  FragU f;
  f.u[0] = *(const uint4*)p;
  f.u[1] = *(const uint4*)(p + 8);
  return f.v;
}

// B operand (32x16 bf16, K x N) where source is row-major [n][k] (k contiguous).
__device__ __forceinline__ v16bf load_b_bf16(const __bf16* base, int n0,
                                             int ld, int k0, int lane) {
  const int sel = (lane >> 4) & 1;
  const __bf16* p = base + (size_t)(n0 + (lane & 15)) * ld + k0 + 16 * sel;
  FragU f;
  f.u[0] = *(const uint4*)p;
  f.u[1] = *(const uint4*)(p + 8);
  return f.v;
}

__device__ __forceinline__ uint4 pack_bf16x8(const v8f& a, float scale) {
  Pack8 t;
#pragma unroll
  for (int i = 0; i < 8; i++) t.h[i] = (__bf16)(a[i] * scale);
  return t.u;
}

// ---------------------------------------------------------------------------
// TDM: DMA a 2D tile (tile_d0 x tile_d1, 8-byte elements, row stride
// stride_d0 in 8B units) from global memory into LDS at lds_off.
// D# layout per CDNA5 ISA ch.8 (group0: count/lds/global/type; group1:
// data_size=8B, tensor dims large (no OOB), tile dims + dim0 stride).
// ---------------------------------------------------------------------------
__device__ __forceinline__ void tdm_load_tile(unsigned lds_off, const void* gsrc,
                                              unsigned tile_d0, unsigned tile_d1,
                                              unsigned stride_d0) {
  const unsigned long long ga = (unsigned long long)(size_t)gsrc;
  v4u g0;
  g0[0] = 1u;                                             // count=1, user D#
  g0[1] = lds_off;                                        // LDS byte address
  g0[2] = (unsigned)ga;                                   // global addr [31:0]
  g0[3] = (unsigned)((ga >> 32) & 0x1FFFFFFu) | (2u << 30); // [56:32] | type=2

  v8i g1;
  g1[0] = (int)(3u << 16);                 // workgroup_mask=0, data_size=3 (8B)
  g1[1] = 0;                               // no atomic barrier; tensor_dim0 lo16=0
  g1[2] = (int)0x10u;                      // tensor_dim0 = 1<<20 ; tensor_dim1 lo16=0
  g1[3] = (int)(0x10u | (tile_d0 << 16));  // tensor_dim1 = 1<<20 ; tile_dim0
  g1[4] = (int)(tile_d1 & 0xFFFFu);        // tile_dim1 ; tile_dim2=0
  g1[5] = (int)stride_d0;                  // tensor_dim0_stride low 32
  g1[6] = 0;
  g1[7] = 0;

  const v4i gz = {0, 0, 0, 0};             // groups 2/3 unused (2D tile)
#if defined(__clang_major__) && __clang_major__ >= 23
  const v8i gz8 = {0, 0, 0, 0, 0, 0, 0, 0};
  __builtin_amdgcn_tensor_load_to_lds(g0, g1, gz, gz, gz8, 0);
#else
  __builtin_amdgcn_tensor_load_to_lds(g0, g1, gz, gz, 0);
#endif
}

// ---------------------------------------------------------------------------
// Kernel 0a: fp32 W[k][n] -> bf16 WT[n][k], LDS-tiled 32x32 transpose.
// ---------------------------------------------------------------------------
__global__ void __launch_bounds__(256) convert_wT(const float* __restrict__ W,
                                                  __bf16* __restrict__ WT) {
  __shared__ float tile[32][33];
  const int tx = threadIdx.x & 31;
  const int ty = threadIdx.x >> 5;          // 0..7
  const int k0 = blockIdx.x * 32;
  const int n0 = blockIdx.y * 32;
#pragma unroll
  for (int i = 0; i < 4; i++)               // coalesced read along n
    tile[ty + 8*i][tx] = W[(size_t)(k0 + ty + 8*i) * D_ + n0 + tx];
  __syncthreads();
#pragma unroll
  for (int i = 0; i < 4; i++)               // coalesced write along k
    WT[(size_t)(n0 + ty + 8*i) * D_ + k0 + tx] = (__bf16)tile[tx][ty + 8*i];
}

// ---------------------------------------------------------------------------
// Kernel 0b: streaming fp32 -> bf16 (8 elements/thread, b128 in/out).
// ---------------------------------------------------------------------------
__global__ void __launch_bounds__(256) convert_bf16(const float* __restrict__ X,
                                                    __bf16* __restrict__ Y) {
  const size_t i = ((size_t)blockIdx.x * 256 + threadIdx.x) * 8;
  const float4 a = *(const float4*)(X + i);
  const float4 b = *(const float4*)(X + i + 4);
  Pack8 p;
  p.h[0] = (__bf16)a.x; p.h[1] = (__bf16)a.y;
  p.h[2] = (__bf16)a.z; p.h[3] = (__bf16)a.w;
  p.h[4] = (__bf16)b.x; p.h[5] = (__bf16)b.y;
  p.h[6] = (__bf16)b.z; p.h[7] = (__bf16)b.w;
  *(uint4*)(Y + i) = p.u;
}

// ---------------------------------------------------------------------------
// Kernel 1: Q/K projection, transposed orientation (M=features, N=tokens).
// A = WT bf16 [n][k], B = Xb bf16 [t][k].  Out: bf16 dst[B][H][S][DH], scaled.
// ---------------------------------------------------------------------------
__global__ void __launch_bounds__(256) proj_qk(const __bf16* __restrict__ Xb,
                                               const __bf16* __restrict__ WT,
                                               const float* __restrict__ bias,
                                               __bf16* __restrict__ dst,
                                               float scale) {
  const int lane = threadIdx.x & 31;
  const int wid  = blockIdx.x * 8 + (threadIdx.x >> 5);  // 0..1023
  const int m0   = (wid & 15) * 64;   // feature tile
  const int t0   = (wid >> 4) * 64;   // token tile

  const v8f z = {0,0,0,0,0,0,0,0};
  v8f acc[4][4];
#pragma unroll
  for (int i = 0; i < 4; i++)
#pragma unroll
    for (int j = 0; j < 4; j++) acc[i][j] = z;

  for (int k0 = 0; k0 < D_; k0 += 32) {
    v16bf a[4], b[4];
#pragma unroll
    for (int mt = 0; mt < 4; mt++) a[mt] = load_a_bf16(WT, m0 + mt*16, D_, k0, lane);
#pragma unroll
    for (int nt = 0; nt < 4; nt++) b[nt] = load_b_bf16(Xb, t0 + nt*16, D_, k0, lane);
#pragma unroll
    for (int mt = 0; mt < 4; mt++)
#pragma unroll
      for (int nt = 0; nt < 4; nt++) acc[mt][nt] = wmma_bf16(a[mt], b[nt], acc[mt][nt]);
  }

  const int lhalf = lane >> 4, l16 = lane & 15;
#pragma unroll
  for (int mt = 0; mt < 4; mt++) {
    const int fbase = m0 + mt*16 + 8*lhalf;       // 8 consecutive features
    const int hh = fbase >> 6, dd = fbase & 63;
    float bv[8];
#pragma unroll
    for (int i = 0; i < 8; i++) bv[i] = bias[fbase + i];
#pragma unroll
    for (int nt = 0; nt < 4; nt++) {
      const int t = t0 + nt*16 + l16;
      const int bb = t >> 11, ss = t & (S_ - 1);
      v8f v = acc[mt][nt];
#pragma unroll
      for (int i = 0; i < 8; i++) v[i] += bv[i];
      __bf16* p = dst + (((size_t)bb * H_ + hh) * S_ + ss) * DH_ + dd;
      *(uint4*)p = pack_bf16x8(v, scale);
    }
  }
}

// ---------------------------------------------------------------------------
// Kernel 2: V projection, natural orientation (M=tokens, N=features).
// A = Xb bf16 [t][k], B = WT bf16 [n][k].  Out: bf16 vT[B][H][DH][S].
// ---------------------------------------------------------------------------
__global__ void __launch_bounds__(256) proj_v(const __bf16* __restrict__ Xb,
                                              const __bf16* __restrict__ WT,
                                              const float* __restrict__ bias,
                                              __bf16* __restrict__ vT) {
  const int lane = threadIdx.x & 31;
  const int wid  = blockIdx.x * 8 + (threadIdx.x >> 5);
  const int m0   = (wid & 63) * 64;   // token tile
  const int n0   = (wid >> 6) * 64;   // feature tile

  const v8f z = {0,0,0,0,0,0,0,0};
  v8f acc[4][4];
#pragma unroll
  for (int i = 0; i < 4; i++)
#pragma unroll
    for (int j = 0; j < 4; j++) acc[i][j] = z;

  for (int k0 = 0; k0 < D_; k0 += 32) {
    v16bf a[4], b[4];
#pragma unroll
    for (int mt = 0; mt < 4; mt++) a[mt] = load_a_bf16(Xb, m0 + mt*16, D_, k0, lane);
#pragma unroll
    for (int nt = 0; nt < 4; nt++) b[nt] = load_b_bf16(WT, n0 + nt*16, D_, k0, lane);
#pragma unroll
    for (int mt = 0; mt < 4; mt++)
#pragma unroll
      for (int nt = 0; nt < 4; nt++) acc[mt][nt] = wmma_bf16(a[mt], b[nt], acc[mt][nt]);
  }

  const int lhalf = lane >> 4, l16 = lane & 15;
#pragma unroll
  for (int mt = 0; mt < 4; mt++) {
    const int tbase = m0 + mt*16 + 8*lhalf;       // 8 consecutive tokens
    const int bb = tbase >> 11, ss = tbase & (S_ - 1);
#pragma unroll
    for (int nt = 0; nt < 4; nt++) {
      const int f = n0 + nt*16 + l16;
      const int hh = f >> 6, dd = f & 63;
      const float bvv = bias[f];
      v8f v = acc[mt][nt];
#pragma unroll
      for (int i = 0; i < 8; i++) v[i] += bvv;
      __bf16* p = vT + (((size_t)bb * H_ + hh) * DH_ + dd) * S_ + ss;
      *(uint4*)p = pack_bf16x8(v, 1.0f);
    }
  }
}

// ---------------------------------------------------------------------------
// Kernel 3: flash attention, transposed-score formulation, TDM-fed.
// One wave per (b,h, 64-query block). The TDM double-buffers the K tile
// (contiguous 8 KB) and V^T tile (64 rows x 128 B, stride 4 KB) into LDS
// (TENSORcnt), overlapping DMA of block j+1 with WMMA/softmax of block j.
// ---------------------------------------------------------------------------
__global__ void __launch_bounds__(32) attn(const __bf16* __restrict__ qh,
                                           const __bf16* __restrict__ kh,
                                           const __bf16* __restrict__ vT,
                                           __bf16* __restrict__ ctx) {
  __shared__ __align__(16) __bf16 Kt[2][64 * 64];   // K tile  [j][d]  x2 buf
  __shared__ __align__(16) __bf16 Vt[2][64 * 64];   // V^T tile [d][j] x2 buf
  __shared__ uint4 Plds[64 * 8];                    // P[q][j] bf16, 8 KB

  const int lane = threadIdx.x & 31;
  const int lhalf = lane >> 4, l16 = lane & 15;
  const int q0 = blockIdx.x * 64;
  const int bh = blockIdx.y;
  const int bb = bh >> 4, hh = bh & 15;

  const __bf16* qp = qh + ((size_t)bb * H_ + hh) * S_ * DH_;
  const __bf16* kp = kh + ((size_t)bb * H_ + hh) * S_ * DH_;
  const __bf16* vp = vT + ((size_t)bb * H_ + hh) * DH_ * S_;

  // Q fragments as B operands (queries are N). Q already scaled by 1/32.
  v16bf bq[2][4];   // [kt][qt]
#pragma unroll
  for (int kt = 0; kt < 2; kt++)
#pragma unroll
    for (int qt = 0; qt < 4; qt++)
      bq[kt][qt] = load_b_bf16(qp, q0 + qt*16, DH_, kt*32, lane);

  const v8f z = {0,0,0,0,0,0,0,0};
  v8f o[4][4];      // O^T frags: [dt][qt], rows=features, cols=queries
#pragma unroll
  for (int i = 0; i < 4; i++)
#pragma unroll
    for (int j = 0; j < 4; j++) o[i][j] = z;

  float mstat[4], lstat[4];
#pragma unroll
  for (int qt = 0; qt < 4; qt++) { mstat[qt] = -1e30f; lstat[qt] = 0.0f; }

  // prologue: DMA key-block 0 into buffer 0
  tdm_load_tile((unsigned)(size_t)&Kt[0][0], kp, /*d0=*/1024, /*d1=*/1, /*stride=*/1024);
  tdm_load_tile((unsigned)(size_t)&Vt[0][0], vp, /*d0=*/16, /*d1=*/64, /*stride=*/(S_ * 2) / 8);

  const int NB = S_ / 64;
  for (int jb = 0; jb < NB; jb++) {
    const int buf = jb & 1;
    const int j0 = jb * 64;

    if (jb + 1 < NB) {
      // DMA next key block into the other buffer, then wait for current block
      // (TDM ops complete in order; allow the 2 just-issued to stay in flight).
      tdm_load_tile((unsigned)(size_t)&Kt[buf ^ 1][0],
                    kp + (size_t)(j0 + 64) * DH_, 1024, 1, 1024);
      tdm_load_tile((unsigned)(size_t)&Vt[buf ^ 1][0],
                    vp + (size_t)(j0 + 64), 16, 64, (S_ * 2) / 8);
      __builtin_amdgcn_s_wait_tensorcnt(2);
    } else {
      __builtin_amdgcn_s_wait_tensorcnt(0);
    }

    const __bf16* kt_l = &Kt[buf][0];
    const __bf16* vt_l = &Vt[buf][0];

    // ---- S^T = K * Q^T : [key][query] ----
    v8f s[4][4];    // [jt][qt]
#pragma unroll
    for (int i = 0; i < 4; i++)
#pragma unroll
      for (int j = 0; j < 4; j++) s[i][j] = z;

#pragma unroll
    for (int jt = 0; jt < 4; jt++) {
      v16bf a0 = load_a_bf16(kt_l, jt*16, DH_, 0, lane);
#pragma unroll
      for (int qt = 0; qt < 4; qt++) s[jt][qt] = wmma_bf16(a0, bq[0][qt], s[jt][qt]);
      v16bf a1 = load_a_bf16(kt_l, jt*16, DH_, 32, lane);
#pragma unroll
      for (int qt = 0; qt < 4; qt++) s[jt][qt] = wmma_bf16(a1, bq[1][qt], s[jt][qt]);
    }

    // ---- online softmax (per query = per lane%16, per qt) ----
#pragma unroll
    for (int qt = 0; qt < 4; qt++) {
      float mloc = -1e30f;
#pragma unroll
      for (int jt = 0; jt < 4; jt++)
#pragma unroll
        for (int i = 0; i < 8; i++) mloc = fmaxf(mloc, s[jt][qt][i]);
      mloc = fmaxf(mloc, __shfl_xor(mloc, 16, 32));     // merge key halves
      const float mnew  = fmaxf(mstat[qt], mloc);
      const float alpha = __expf(mstat[qt] - mnew);
      mstat[qt] = mnew;

      float rsum = 0.0f;
#pragma unroll
      for (int jt = 0; jt < 4; jt++) {
#pragma unroll
        for (int i = 0; i < 8; i++) {
          const float e = __expf(s[jt][qt][i] - mnew);
          s[jt][qt][i] = e;
          rsum += e;
        }
      }
      rsum += __shfl_xor(rsum, 16, 32);
      lstat[qt] = lstat[qt] * alpha + rsum;

#pragma unroll
      for (int dt = 0; dt < 4; dt++)
#pragma unroll
        for (int i = 0; i < 8; i++) o[dt][qt][i] *= alpha;

      // store P (bf16) to LDS as [q][j], j contiguous: one b128 per frag
#pragma unroll
      for (int jt = 0; jt < 4; jt++)
        Plds[(qt*16 + l16) * 8 + jt*2 + lhalf] = pack_bf16x8(s[jt][qt], 1.0f);
    }

    // ---- O^T += V^T * P^T ----
#pragma unroll
    for (int kt2 = 0; kt2 < 2; kt2++) {
      v16bf bp[4];
#pragma unroll
      for (int qt = 0; qt < 4; qt++) {
        const int base = (qt*16 + l16) * 8 + kt2*4 + 2*lhalf;
        FragU f;
        f.u[0] = Plds[base];
        f.u[1] = Plds[base + 1];
        bp[qt] = f.v;
      }
#pragma unroll
      for (int dt = 0; dt < 4; dt++) {
        v16bf av = load_a_bf16(vt_l, dt*16, DH_, kt2*32, lane);
#pragma unroll
        for (int qt = 0; qt < 4; qt++) o[dt][qt] = wmma_bf16(av, bp[qt], o[dt][qt]);
      }
    }
  }

  // ---- epilogue: O = O^T / l, write ctx[b][s][h*64+d] (bf16) ----
#pragma unroll
  for (int qt = 0; qt < 4; qt++) {
    const float inv = 1.0f / lstat[qt];
    const int t = q0 + qt*16 + l16;
#pragma unroll
    for (int dt = 0; dt < 4; dt++) {
      __bf16* p = ctx + ((size_t)bb * S_ + t) * D_ + hh * DH_ + dt*16 + 8*lhalf;
      *(uint4*)p = pack_bf16x8(o[dt][qt], inv);
    }
  }
}

// ---------------------------------------------------------------------------
// Kernel 4: output projection, transposed orientation.
// A = WoT bf16 [n][k], B = ctx bf16 [t][k].  Out: fp32 d_out[t][n] + bias.
// ---------------------------------------------------------------------------
__global__ void __launch_bounds__(256) out_proj(const __bf16* __restrict__ CTX,
                                                const __bf16* __restrict__ WT,
                                                const float* __restrict__ bias,
                                                float* __restrict__ out) {
  const int lane = threadIdx.x & 31;
  const int wid  = blockIdx.x * 8 + (threadIdx.x >> 5);
  const int m0   = (wid & 15) * 64;   // feature tile
  const int t0   = (wid >> 4) * 64;   // token tile

  const v8f z = {0,0,0,0,0,0,0,0};
  v8f acc[4][4];
#pragma unroll
  for (int i = 0; i < 4; i++)
#pragma unroll
    for (int j = 0; j < 4; j++) acc[i][j] = z;

  for (int k0 = 0; k0 < D_; k0 += 32) {
    v16bf a[4], b[4];
#pragma unroll
    for (int mt = 0; mt < 4; mt++) a[mt] = load_a_bf16(WT, m0 + mt*16, D_, k0, lane);
#pragma unroll
    for (int nt = 0; nt < 4; nt++) b[nt] = load_b_bf16(CTX, t0 + nt*16, D_, k0, lane);
#pragma unroll
    for (int mt = 0; mt < 4; mt++)
#pragma unroll
      for (int nt = 0; nt < 4; nt++) acc[mt][nt] = wmma_bf16(a[mt], b[nt], acc[mt][nt]);
  }

  const int lhalf = lane >> 4, l16 = lane & 15;
#pragma unroll
  for (int mt = 0; mt < 4; mt++) {
    const int fbase = m0 + mt*16 + 8*lhalf;
    float bv[8];
#pragma unroll
    for (int i = 0; i < 8; i++) bv[i] = bias[fbase + i];
#pragma unroll
    for (int nt = 0; nt < 4; nt++) {
      const int t = t0 + nt*16 + l16;
      float* p = out + (size_t)t * D_ + fbase;
      float4 o0 = { acc[mt][nt][0] + bv[0], acc[mt][nt][1] + bv[1],
                    acc[mt][nt][2] + bv[2], acc[mt][nt][3] + bv[3] };
      float4 o1 = { acc[mt][nt][4] + bv[4], acc[mt][nt][5] + bv[5],
                    acc[mt][nt][6] + bv[6], acc[mt][nt][7] + bv[7] };
      *(float4*)p       = o0;
      *(float4*)(p + 4) = o1;
    }
  }
}

// ---------------------------------------------------------------------------
extern "C" void kernel_launch(void* const* d_in, const int* in_sizes, int n_in,
                              void* d_out, int out_size, void* d_ws, size_t ws_size,
                              hipStream_t stream) {
  (void)in_sizes; (void)n_in; (void)out_size; (void)ws_size;

  const float* V  = (const float*)d_in[0];
  const float* Q  = (const float*)d_in[1];
  const float* K  = (const float*)d_in[2];
  const float* Wq = (const float*)d_in[3];
  const float* bq = (const float*)d_in[4];
  const float* Wk = (const float*)d_in[5];
  const float* bk = (const float*)d_in[6];
  const float* Wv = (const float*)d_in[7];
  const float* bv = (const float*)d_in[8];
  const float* Wo = (const float*)d_in[9];
  const float* bo = (const float*)d_in[10];
  float* out = (float*)d_out;

  // Workspace layout (peak 56 MB):
  //   [ 0, 8MB)   WqT WkT WvT WoT (bf16)
  //   [ 8,32MB)   qh kh vTh (bf16, head-split; vT is [b][h][dh][s])
  //   [32,56MB)   phase 1: Qb Kb Vb (bf16 activations)
  //   [32,40MB)   phase 2: ctx (bf16)  -- reuses Qb region after projections
  char* ws = (char*)d_ws;
  const size_t WBYTES = (size_t)D_ * D_ * 2;                 // 2 MB per weight
  const size_t HBYTES = (size_t)B_ * H_ * S_ * DH_ * 2;      // 8 MB per tensor
  __bf16* WqT = (__bf16*)(ws + 0 * WBYTES);
  __bf16* WkT = (__bf16*)(ws + 1 * WBYTES);
  __bf16* WvT = (__bf16*)(ws + 2 * WBYTES);
  __bf16* WoT = (__bf16*)(ws + 3 * WBYTES);
  __bf16* qh  = (__bf16*)(ws + 4 * WBYTES);
  __bf16* kh  = (__bf16*)(ws + 4 * WBYTES + 1 * HBYTES);
  __bf16* vTh = (__bf16*)(ws + 4 * WBYTES + 2 * HBYTES);
  __bf16* Qb  = (__bf16*)(ws + 4 * WBYTES + 3 * HBYTES);
  __bf16* Kb  = (__bf16*)(ws + 4 * WBYTES + 4 * HBYTES);
  __bf16* Vb  = (__bf16*)(ws + 4 * WBYTES + 5 * HBYTES);
  __bf16* ctx = Qb;   // safe alias: Qb dead before attn writes ctx

  // 0a) weight transpose+convert (coalesced via LDS tiles)
  dim3 tg(D_ / 32, D_ / 32);
  convert_wT<<<tg, 256, 0, stream>>>(Wq, WqT);
  convert_wT<<<tg, 256, 0, stream>>>(Wk, WkT);
  convert_wT<<<tg, 256, 0, stream>>>(Wv, WvT);
  convert_wT<<<tg, 256, 0, stream>>>(Wo, WoT);

  // 0b) activation convert (8 elem/thread)
  const int cvt_blocks = (T_ * D_) / (256 * 8);
  convert_bf16<<<cvt_blocks, 256, 0, stream>>>(Q, Qb);
  convert_bf16<<<cvt_blocks, 256, 0, stream>>>(K, Kb);
  convert_bf16<<<cvt_blocks, 256, 0, stream>>>(V, Vb);

  // 1-2) projections; scores' 1/sqrt(KEY_DIM)=1/32 folded into Q projection
  proj_qk<<<128, 256, 0, stream>>>(Qb, WqT, bq, qh, 0.03125f);
  proj_qk<<<128, 256, 0, stream>>>(Kb, WkT, bk, kh, 1.0f);
  proj_v <<<128, 256, 0, stream>>>(Vb, WvT, bv, vTh);

  // 3) attention (TDM double-buffered K/V)
  attn<<<dim3(S_ / 64, B_ * H_), 32, 0, stream>>>(qh, kh, vTh, ctx);

  // 4) output projection
  out_proj<<<128, 256, 0, stream>>>(ctx, WoT, bo, out);
}
